// DecoderWithTemporalAttention_47485158424752
// MI455X (gfx1250) — matrix operand order repaired and checked by reading, not compiled, not run on hardware
//
#include <hip/hip_runtime.h>
#include <hip/hip_bf16.h>
#include <stdint.h>

#define B_   256
#define T_   512
#define HE_  256
#define M_   (B_ * T_)      // 131072 rows of the fused attention GEMM

typedef __attribute__((ext_vector_type(16))) __bf16 v16bf;
typedef __attribute__((ext_vector_type(8)))  float  v8f;
typedef __attribute__((ext_vector_type(4)))  unsigned int u32x4;
typedef __attribute__((ext_vector_type(8)))  int          i32x8;
typedef __attribute__((ext_vector_type(4)))  int          i32x4;

#if __has_builtin(__builtin_amdgcn_tensor_load_to_lds) && \
    __has_builtin(__builtin_amdgcn_s_wait_tensorcnt)
#define USE_TDM 1
#endif

static __device__ __forceinline__ unsigned short f32_to_bf16_rne(float f) {
    union { float f; unsigned u; } v; v.f = f;
    unsigned rounding = 0x7FFFu + ((v.u >> 16) & 1u);
    return (unsigned short)((v.u + rounding) >> 16);
}

union AFrag { v16bf v; unsigned short us[16]; };
union BFrag { v16bf v; uint4 q[2]; };

// ---------------------------------------------------------------------------
// Kernel 1: Ud_w fp32 -> bf16 (one-time, 128 KB)
// ---------------------------------------------------------------------------
__global__ __launch_bounds__(256) void cvt_bf16_kernel(const float* __restrict__ src,
                                                       unsigned short* __restrict__ dst,
                                                       int n) {
    int i = blockIdx.x * 256 + threadIdx.x;
    if (i < n) dst[i] = f32_to_bf16_rne(src[i]);
}

// ---------------------------------------------------------------------------
// Kernel 2: fused  scores[m] = vd_w . tanh(Wd_b + Ud_b + Ud_w @ h[m]) + vd_b
//   GEMM  M=131072, N=256, K=256  with v_wmma_f32_16x16x32_bf16.
//   Block = 256 threads = 8 wave32 waves, block tile = 128 rows x 256 cols.
//   Full B (Ud_w bf16, 128 KB) staged in LDS once per block via the
//   Tensor Data Mover (tensor_load_to_lds + s_wait_tensorcnt).
// ---------------------------------------------------------------------------
__global__ __launch_bounds__(256) void attn_scores_kernel(
    const float* __restrict__ h,             // [M_][HE_] fp32 (row-major, K contiguous)
    const unsigned short* __restrict__ Ubf,  // [HE_][HE_] bf16, Ud_w layout (n, k)
    const float* __restrict__ Wd_b,
    const float* __restrict__ Ud_b,
    const float* __restrict__ vd_w,
    const float* __restrict__ vd_b,
    float* __restrict__ scores)              // [M_]
{
    extern __shared__ unsigned short Bs[];   // HE_*HE_ bf16 = 128 KB

    const int tid = threadIdx.x;

#ifdef USE_TDM
    // One TDM descriptor DMAs the whole 256x256 bf16 B matrix into LDS.
    // TDM ignores EXEC and issues per-wave: let only wave 0 issue it.
    if (tid < 32) {
        const unsigned lds_base = (unsigned)(uintptr_t)(&Bs[0]);
        const unsigned long long ga = (unsigned long long)(uintptr_t)Ubf;
        // D# group 0: count=1 | lds_addr | global_addr[56:0] | type=2
        u32x4 g0 = { 1u,
                     lds_base,
                     (unsigned)(ga & 0xFFFFFFFFu),
                     (unsigned)(ga >> 32) | (2u << 30) };
        // D# group 1: workgroup_mask=0, data_size=1 (2B), no pad/iterate,
        //   tensor_dim0=256, tensor_dim1=256, tile_dim0=256, tile_dim1=256,
        //   tensor_dim0_stride=256, tensor_dim1_stride=0 (2-D tile)
        i32x8 g1 = { (int)0x00010000u,        // data_size=2B
                     (int)(256u << 16),       // tensor_dim0[15:0] in [31:16]
                     (int)(256u << 16),       // tensor_dim1[15:0] in [31:16]
                     (int)(256u << 16),       // tile_dim0 in [31:16]
                     (int)256,                // tile_dim1=256, tile_dim2=0
                     (int)256,                // tensor_dim0_stride[31:0]
                     0, 0 };
        i32x4 gz = { 0, 0, 0, 0 };            // groups 2/3 unused (2-D tensor)
        i32x8 gz8 = { 0, 0, 0, 0, 0, 0, 0, 0 };
        __builtin_amdgcn_tensor_load_to_lds(g0, g1, gz, gz, gz8, 0);
        __builtin_amdgcn_s_wait_tensorcnt(0);
    }
#else
    // Fallback: cooperative staged copy, 8192 x 16B, coalesced.
    {
        const uint4* gsrc = (const uint4*)Ubf;
        uint4*       ldst = (uint4*)Bs;
        #pragma unroll
        for (int i = 0; i < 32; ++i)
            ldst[tid + 256 * i] = gsrc[tid + 256 * i];
    }
#endif
    __syncthreads();

    const int wave = tid >> 5;           // wave32
    const int lane = tid & 31;
    const int l16  = lane & 15;
    const int hsel = lane >> 4;          // lane half: 0 or 1
    const size_t mbase = (size_t)blockIdx.x * 128 + (size_t)wave * 16;

    v8f c[16];
    #pragma unroll
    for (int t = 0; t < 16; ++t) c[t] = (v8f){0.f,0.f,0.f,0.f,0.f,0.f,0.f,0.f};

    // A-fragment (16-bit 16x32) lane layout:
    //   lanes 0-15 (M=l16):  K = kc+0..7  (V0-3) and kc+16..23 (V4-7)
    //   lanes 16-31(M=l16):  K = kc+8..15 (V0-3) and kc+24..31 (V4-7)
    const float* arow = h + (mbase + (size_t)l16) * HE_ + (hsel << 3);

    #pragma unroll 1
    for (int kc = 0; kc < HE_; kc += 32) {
        __builtin_prefetch(arow + kc + 64, 0, 1);   // global_prefetch_b8 for A stream

        union { float4 f4[4]; float f[16]; } af;
        af.f4[0] = *(const float4*)(arow + kc);
        af.f4[1] = *(const float4*)(arow + kc + 4);
        af.f4[2] = *(const float4*)(arow + kc + 16);
        af.f4[3] = *(const float4*)(arow + kc + 20);

        AFrag a;
        #pragma unroll
        for (int j = 0; j < 16; ++j) a.us[j] = f32_to_bf16_rne(af.f[j]);

        // B-fragment (32x16 bf16): lane reads k = kc+8h..+7 and kc+16+8h..+7 of Ud_w row n.
        const int kbB = kc + (hsel << 3);
        #pragma unroll
        for (int ti = 0; ti < 16; ++ti) {
            BFrag b;
            const uint4* bp = (const uint4*)(Bs + ((ti * 16 + l16) * HE_ + kbB));
            b.q[0] = bp[0];
            b.q[1] = bp[2];
            c[ti] = __builtin_amdgcn_wmma_f32_16x16x32_bf16(
                        /*neg_a=*/false, a.v, /*neg_b=*/false, b.v,
                        /*c_mod=*/(short)0, c[ti],
                        /*reuse_a=*/false, /*reuse_b=*/false);
        }
    }

    // Epilogue: tanh(+bias) * vd_w, reduce across N (lanes within each 16-half).
    // C layout: c[ti] vgpr r, lane l -> row mbase + r + (l<16?0:8), col ti*16 + (l&15).
    float sacc[8] = {0.f,0.f,0.f,0.f,0.f,0.f,0.f,0.f};
    #pragma unroll
    for (int ti = 0; ti < 16; ++ti) {
        const int n     = ti * 16 + l16;
        const float bia = Wd_b[n] + Ud_b[n];
        const float vw  = vd_w[n];
        #pragma unroll
        for (int r = 0; r < 8; ++r)
            sacc[r] += tanhf(c[ti][r] + bia) * vw;
    }
    #pragma unroll
    for (int off = 8; off >= 1; off >>= 1) {
        #pragma unroll
        for (int r = 0; r < 8; ++r)
            sacc[r] += __shfl_xor(sacc[r], off, 32);   // stays within each 16-lane half
    }
    if (l16 == 0) {
        const float vb = vd_b[0];
        const size_t rb = mbase + (hsel << 3);
        #pragma unroll
        for (int r = 0; r < 8; ++r) scores[rb + r] = sacc[r] + vb;
    }
}

// ---------------------------------------------------------------------------
// Kernel 3: per-batch softmax over T, c_t = beta @ h, scalarized LSTM cell.
//   out[0 .. 65535]      = d_new  (B_ x HD)
//   out[65536 .. 131071] = c_t    (B_ x HE)
// ---------------------------------------------------------------------------
__global__ __launch_bounds__(256) void softmax_ct_lstm_kernel(
    const float* __restrict__ scores,
    const float* __restrict__ h,
    const float* __restrict__ y,        // (B, T, 1)
    const float* __restrict__ wt_w,     // (1, HE+1)
    const float* __restrict__ wt_b,
    const float* __restrict__ W_ih,     // (4*HD, 1) flat
    const float* __restrict__ b_ih,
    const float* __restrict__ b_hh,
    float* __restrict__ out)
{
    __shared__ float beta[T_];
    __shared__ float red[256];
    const int b   = blockIdx.x;
    const int tid = threadIdx.x;

    const float s0v = scores[b * T_ + tid];
    const float s1v = scores[b * T_ + 256 + tid];

    // block max
    red[tid] = fmaxf(s0v, s1v);
    __syncthreads();
    for (int off = 128; off > 0; off >>= 1) {
        if (tid < off) red[tid] = fmaxf(red[tid], red[tid + off]);
        __syncthreads();
    }
    const float mx = red[0];
    __syncthreads();

    const float e0 = expf(s0v - mx);
    const float e1 = expf(s1v - mx);

    // block sum
    red[tid] = e0 + e1;
    __syncthreads();
    for (int off = 128; off > 0; off >>= 1) {
        if (tid < off) red[tid] += red[tid + off];
        __syncthreads();
    }
    const float inv = 1.0f / red[0];
    __syncthreads();

    beta[tid]       = e0 * inv;
    beta[tid + 256] = e1 * inv;
    __syncthreads();

    // c_t[b, tid] = sum_t beta[t] * h[b, t, tid]   (coalesced over tid)
    float acc = 0.f;
    const float* hb = h + (size_t)b * T_ * HE_ + tid;
    #pragma unroll 4
    for (int t = 0; t < T_; ++t)
        acc = fmaf(beta[t], hb[(size_t)t * HE_], acc);
    out[(size_t)(B_ * HE_) + (size_t)b * HE_ + tid] = acc;   // c_t output

    // y_tilde = c_t . wt_w[0:HE] + y_t * wt_w[HE] + wt_b   (scalar per batch)
    red[tid] = acc * wt_w[tid];
    __syncthreads();
    for (int off = 128; off > 0; off >>= 1) {
        if (tid < off) red[tid] += red[tid + off];
        __syncthreads();
    }
    const float ytld = red[0] + y[(size_t)b * T_] * wt_w[HE_] + wt_b[0];

    // gates (d0 == 0, s0 == 0 -> f-gate dead, c_new = i*g)
    const float gi = ytld * W_ih[tid]       + b_ih[tid]       + b_hh[tid];
    const float gg = ytld * W_ih[512 + tid] + b_ih[512 + tid] + b_hh[512 + tid];
    const float go = ytld * W_ih[768 + tid] + b_ih[768 + tid] + b_hh[768 + tid];
    const float iv = 1.f / (1.f + expf(-gi));
    const float gv = tanhf(gg);
    const float ov = 1.f / (1.f + expf(-go));
    out[(size_t)b * HE_ + tid] = ov * tanhf(iv * gv);        // d_new output
}

// ---------------------------------------------------------------------------
extern "C" void kernel_launch(void* const* d_in, const int* in_sizes, int n_in,
                              void* d_out, int out_size, void* d_ws, size_t ws_size,
                              hipStream_t stream) {
    (void)in_sizes; (void)n_in; (void)out_size; (void)ws_size;

    const float* y    = (const float*)d_in[0];
    const float* h    = (const float*)d_in[1];
    // d_in[2] (Wd_w) unused: multiplied by zero state
    const float* Wd_b = (const float*)d_in[3];
    const float* Ud_w = (const float*)d_in[4];
    const float* Ud_b = (const float*)d_in[5];
    const float* vd_w = (const float*)d_in[6];
    const float* vd_b = (const float*)d_in[7];
    const float* wt_w = (const float*)d_in[8];
    const float* wt_b = (const float*)d_in[9];
    const float* W_ih = (const float*)d_in[10];
    // d_in[11] (W_hh) unused: multiplied by zero state
    const float* b_ih = (const float*)d_in[12];
    const float* b_hh = (const float*)d_in[13];
    float* out = (float*)d_out;

    // workspace layout: scores (M_ floats = 512 KB) | Ud_w bf16 (128 KB)
    float* scores       = (float*)d_ws;
    unsigned short* Ubf = (unsigned short*)((char*)d_ws + (size_t)M_ * sizeof(float));

    (void)hipFuncSetAttribute(reinterpret_cast<const void*>(attn_scores_kernel),
                              hipFuncAttributeMaxDynamicSharedMemorySize,
                              HE_ * HE_ * 2);

    cvt_bf16_kernel<<<(HE_ * HE_) / 256, 256, 0, stream>>>(Ud_w, Ubf, HE_ * HE_);
    attn_scores_kernel<<<M_ / 128, 256, HE_ * HE_ * 2, stream>>>(
        h, Ubf, Wd_b, Ud_b, vd_w, vd_b, scores);
    softmax_ct_lstm_kernel<<<B_, 256, 0, stream>>>(
        scores, h, y, wt_w, wt_b, W_ih, b_ih, b_hh, out);
}
